// Rct_32418413150429
// MI455X (gfx1250) — matrix operand order repaired and checked by reading, not compiled, not run on hardware
//
#include <hip/hip_runtime.h>
#include <hip/hip_bf16.h>
#include <cstdint>
#include <cstddef>

// ---------------------------------------------------------------------------
// Types for CDNA5 WMMA (wave32): v_wmma_f32_16x16x32_bf16
//   A: 16x32 bf16 (16 halves / lane), B: 32x16 bf16, C/D: 16x16 f32 (8 / lane)
// ---------------------------------------------------------------------------
typedef __attribute__((ext_vector_type(16))) __bf16 v16bf;
typedef __attribute__((ext_vector_type(8)))  float  v8f;
typedef int v4i __attribute__((vector_size(16)));   // matches builtin param type

union FragBF {
    uint4 q[2];
    v16bf v;
};

__device__ __forceinline__ v8f wmma_bf16(v16bf a, v16bf b, v8f c) {
    return __builtin_amdgcn_wmma_f32_16x16x32_bf16(
        /*neg_a=*/false, a, /*neg_b=*/false, b,
        /*c_mod=*/(short)0, c, /*reuse_a=*/false, /*reuse_b=*/false);
}

// ---------------------------------------------------------------------------
// CDNA5 async LDS-DMA helpers (GLOBAL_LOAD_ASYNC_TO_LDS_B128, ASYNCcnt).
// Builtin signature (from hipcc diagnostic): (v4i AS1*, v4i AS3*, imm, imm).
// ---------------------------------------------------------------------------
#define LDS_AS __attribute__((address_space(3)))
#define GLB_AS __attribute__((address_space(1)))

#if defined(__has_builtin) && __has_builtin(__builtin_amdgcn_global_load_async_to_lds_b128)
#define HAVE_ASYNC_LDS 1
#else
#define HAVE_ASYNC_LDS 0
#endif

__device__ __forceinline__ void async_copy_b128(const __bf16* gsrc, __bf16* lds_dst) {
#if HAVE_ASYNC_LDS
    __builtin_amdgcn_global_load_async_to_lds_b128(
        (GLB_AS v4i*)(uintptr_t)gsrc,                 // global VA == flat VA
        (LDS_AS v4i*)(LDS_AS __bf16*)lds_dst,         // generic -> local addrspacecast
        /*imm offset=*/0, /*cpol=*/0);
#else
    *(uint4*)lds_dst = *(const uint4*)gsrc;
#endif
}

__device__ __forceinline__ void wait_async_lds() {
#if HAVE_ASYNC_LDS
  #if __has_builtin(__builtin_amdgcn_s_wait_asynccnt)
    __builtin_amdgcn_s_wait_asynccnt(0);
  #else
    asm volatile("s_wait_asynccnt 0x0" ::: "memory");
  #endif
#endif
}

// ---------------------------------------------------------------------------
// Problem constants
// ---------------------------------------------------------------------------
#define BATCH   256
#define SIZE    512
#define UNITS   2048     // 4*SIZE
#define GATES4  8192     // 4*UNITS
#define CCH     1024
#define LRELU_A 0.2f
#define EPS_IN  1e-6f

// ===========================================================================
// Generic GEMM: C[M,N] = A[M,K] * B[K,N], A/B fp32 row-major (converted to
// bf16 while staging to LDS), C fp32.  Tiles: 128x128x32, 256 thr = 8 waves.
// Wave (wm in 0..1, wn in 0..3) computes 64x32 = 4x2 WMMA accumulators.
// Staging is two-phase (batch loads, then convert+store) so the loads issue
// as one clause with a single s_wait_loadcnt.
// ===========================================================================
#define GM_BM 128
#define GM_BN 128
#define GM_BK 32
#define GM_LD 40   // padded K stride (halves)

__global__ __launch_bounds__(256) void gemm_f32_bf16wmma(
    const float* __restrict__ A, const float* __restrict__ B,
    float* __restrict__ C, int M, int N, int K)
{
    __shared__ __bf16 a_lds[GM_BM][GM_LD];
    __shared__ __bf16 b_lds[GM_BN][GM_LD];   // transposed: [n][k]

    const int tid   = threadIdx.x;
    const int lane  = tid & 31;
    const int wave  = tid >> 5;
    const int wm    = wave >> 2;      // 0..1 -> 64 rows
    const int wn    = wave & 3;       // 0..3 -> 32 cols
    const int lhalf = lane >> 4;      // 0/1
    const int l15   = lane & 15;

    const int m0 = blockIdx.x * GM_BM;
    const int n0 = blockIdx.y * GM_BN;

    v8f acc[4][2] = {};

    for (int kb = 0; kb < K; kb += GM_BK) {
        // ---- phase 1: batched global loads (A: 4x b128, B: 4x b128) ----
        const int ra = tid >> 3;               // 0..31
        const int ca = (tid & 7) * 4;          // 0..28
        const int kr = tid >> 5;               // 0..7
        const int nc = (tid & 31) * 4;         // 0..124
        float4 av[4], bv[4];
#pragma unroll
        for (int p = 0; p < 4; ++p)
            av[p] = *(const float4*)&A[(size_t)(m0 + ra + p * 32) * K + kb + ca];
#pragma unroll
        for (int p = 0; p < 4; ++p)
            bv[p] = *(const float4*)&B[(size_t)(kb + kr + p * 8) * N + n0 + nc];
        // one 128B-line prefetch per 8 threads covers next B tile exactly
        if ((tid & 7) == 0 && kb + GM_BK < K) {
#pragma unroll
            for (int p = 0; p < 4; ++p)
                __builtin_prefetch(&B[(size_t)(kb + GM_BK + kr + p * 8) * N + n0 + nc], 0, 1);
        }

        __syncthreads();   // previous tile fully consumed
        // ---- phase 2: convert + store to LDS ----
#pragma unroll
        for (int p = 0; p < 4; ++p) {
            a_lds[ra + p * 32][ca + 0] = (__bf16)av[p].x;
            a_lds[ra + p * 32][ca + 1] = (__bf16)av[p].y;
            a_lds[ra + p * 32][ca + 2] = (__bf16)av[p].z;
            a_lds[ra + p * 32][ca + 3] = (__bf16)av[p].w;
        }
#pragma unroll
        for (int p = 0; p < 4; ++p) {
            b_lds[nc + 0][kr + p * 8] = (__bf16)bv[p].x;
            b_lds[nc + 1][kr + p * 8] = (__bf16)bv[p].y;
            b_lds[nc + 2][kr + p * 8] = (__bf16)bv[p].z;
            b_lds[nc + 3][kr + p * 8] = (__bf16)bv[p].w;
        }
        __syncthreads();

        // ---- fragments + WMMA ----
        v16bf afrag[4], bfrag[2];
#pragma unroll
        for (int ms = 0; ms < 4; ++ms) {
            const int m = wm * 64 + ms * 16 + l15;
            FragBF u;
            u.q[0] = *(const uint4*)&a_lds[m][lhalf * 8];
            u.q[1] = *(const uint4*)&a_lds[m][16 + lhalf * 8];
            afrag[ms] = u.v;
        }
#pragma unroll
        for (int ns = 0; ns < 2; ++ns) {
            const int n = wn * 32 + ns * 16 + l15;
            FragBF u;
            u.q[0] = *(const uint4*)&b_lds[n][lhalf * 8];
            u.q[1] = *(const uint4*)&b_lds[n][16 + lhalf * 8];
            bfrag[ns] = u.v;
        }
#pragma unroll
        for (int ms = 0; ms < 4; ++ms)
#pragma unroll
            for (int ns = 0; ns < 2; ++ns)
                acc[ms][ns] = wmma_bf16(afrag[ms], bfrag[ns], acc[ms][ns]);
    }

    // ---- store C ----
#pragma unroll
    for (int ms = 0; ms < 4; ++ms) {
#pragma unroll
        for (int ns = 0; ns < 2; ++ns) {
            const int col = n0 + wn * 32 + ns * 16 + l15;
#pragma unroll
            for (int v = 0; v < 8; ++v) {
                const int row = m0 + wm * 64 + ms * 16 + lhalf * 8 + v;
                C[(size_t)row * N + col] = acc[ms][ns][v];
            }
        }
    }
}

// ===========================================================================
// Fused LSTM cell: z = xt@K + h@R + b ; all gates tanh ; c' = f*c + i*g ;
// h' = o*tanh(c').  WG tile: 64 rows x 64 cols *per gate*, so the epilogue
// has i/f/g/o for the same (row, col) in registers.
// Grid (4, 32), 256 threads = 8 waves; wave = (wm: 32 rows, wn: 16 cols).
// K loop: 64 iters over xt@K then 64 iters over h@R (K-tile = 32).
// A tiles (bf16, no conversion) use GLOBAL_LOAD_ASYNC_TO_LDS_B128 (ASYNCcnt);
// weight tiles: batched b128 loads -> v_cvt_pk_bf16 -> transposed LDS stores,
// with one 128B prefetch per line of the K+2 tile.
// ===========================================================================
#define LC_BM 64
#define LC_BN 64
#define LC_LD 40

__global__ __launch_bounds__(256) void lstm_cell_kernel(
    const __bf16* __restrict__ xt, int x_stride,
    const __bf16* __restrict__ hprev,
    const float* __restrict__ Kw, const float* __restrict__ Rw,
    const float* __restrict__ bias,
    float* __restrict__ c_state,          // in/out [256,2048]
    __bf16* __restrict__ hnew,            // out    [256,2048] bf16
    float* __restrict__ y_out, int w_idx) // optional decoder output scatter
{
    __shared__ __bf16 a_lds[LC_BM][LC_LD];
    __shared__ __bf16 b_lds[4][LC_BN][LC_LD];   // [gate][n][k]

    const int tid   = threadIdx.x;
    const int lane  = tid & 31;
    const int wave  = tid >> 5;
    const int wm    = wave >> 2;      // 0..1 -> 32 rows
    const int wn    = wave & 3;       // 0..3 -> 16 cols
    const int lhalf = lane >> 4;
    const int l15   = lane & 15;

    const int m0 = blockIdx.x * LC_BM;     // 0..192
    const int n0 = blockIdx.y * LC_BN;     // 0..1984 (within 2048)

    v8f acc[4][2] = {};                    // [gate][msub]

    for (int kt = 0; kt < 128; ++kt) {
        const int kb = (kt & 63) * 32;
        const __bf16* Asrc = (kt < 64) ? xt : hprev;
        const int     astr = (kt < 64) ? x_stride : UNITS;
        const float*  Wsrc = (kt < 64) ? Kw : Rw;

        // ---- phase 1: batched weight loads (8x b128 per thread) ----
        float4 wv[8];
#pragma unroll
        for (int it = 0; it < 8; ++it) {
            const int i  = it * 256 + tid;       // 0..2047 float4s
            const int g  = i >> 9;               // gate
            const int rm = i & 511;
            const int kr = rm >> 4;              // 0..31
            const int nc = (rm & 15) * 4;        // 0..60
            wv[it] = *(const float4*)&Wsrc[(size_t)(kb + kr) * GATES4 +
                                           g * UNITS + n0 + nc];
        }
        // one 128B-line prefetch per 8 threads: 256 lines == the K+2 tile
        if ((tid & 7) == 0 && (kt & 63) < 62) {
#pragma unroll
            for (int it = 0; it < 8; ++it) {
                const int i  = it * 256 + tid;
                const int g  = i >> 9;
                const int rm = i & 511;
                const int kr = rm >> 4;
                const int nc = (rm & 15) * 4;
                __builtin_prefetch(&Wsrc[(size_t)(kb + 64 + kr) * GATES4 +
                                         g * UNITS + n0 + nc], 0, 1);
            }
        }

        __syncthreads();   // previous tile fully consumed
        // ---- stage A: 64 rows x 32 halves, one async b128 per thread ----
        {
            const int r  = tid >> 2;        // 0..63
            const int c8 = (tid & 3) * 8;   // 0,8,16,24
            async_copy_b128(&Asrc[(size_t)(m0 + r) * astr + kb + c8],
                            &a_lds[r][c8]);
        }
        // ---- phase 2: convert + transposed store of weight tile ----
#pragma unroll
        for (int it = 0; it < 8; ++it) {
            const int i  = it * 256 + tid;
            const int g  = i >> 9;
            const int rm = i & 511;
            const int kr = rm >> 4;
            const int nc = (rm & 15) * 4;
            b_lds[g][nc + 0][kr] = (__bf16)wv[it].x;
            b_lds[g][nc + 1][kr] = (__bf16)wv[it].y;
            b_lds[g][nc + 2][kr] = (__bf16)wv[it].z;
            b_lds[g][nc + 3][kr] = (__bf16)wv[it].w;
        }
        wait_async_lds();  // ASYNCcnt == 0: A tile landed in LDS
        __syncthreads();   // all waves' B stores + async A visible

        v16bf afrag[2];
#pragma unroll
        for (int ms = 0; ms < 2; ++ms) {
            const int m = wm * 32 + ms * 16 + l15;
            FragBF u;
            u.q[0] = *(const uint4*)&a_lds[m][lhalf * 8];
            u.q[1] = *(const uint4*)&a_lds[m][16 + lhalf * 8];
            afrag[ms] = u.v;
        }
#pragma unroll
        for (int g = 0; g < 4; ++g) {
            const int n = wn * 16 + l15;
            FragBF u;
            u.q[0] = *(const uint4*)&b_lds[g][n][lhalf * 8];
            u.q[1] = *(const uint4*)&b_lds[g][n][16 + lhalf * 8];
            v16bf bfrag = u.v;
#pragma unroll
            for (int ms = 0; ms < 2; ++ms)
                acc[g][ms] = wmma_bf16(afrag[ms], bfrag, acc[g][ms]);
        }
    }

    // ---- epilogue: gates + state update ----
    const int col = n0 + wn * 16 + l15;   // 0..2047
    const float bi = bias[col];
    const float bf = bias[UNITS + col];
    const float bg = bias[2 * UNITS + col];
    const float bo = bias[3 * UNITS + col];

#pragma unroll
    for (int ms = 0; ms < 2; ++ms) {
        const int rbase = m0 + wm * 32 + ms * 16 + lhalf * 8;
#pragma unroll
        for (int v = 0; v < 8; ++v) {
            const int row = rbase + v;
            const float i_ = tanhf(acc[0][ms][v] + bi);
            const float f_ = tanhf(acc[1][ms][v] + bf);
            const float g_ = tanhf(acc[2][ms][v] + bg);
            const float o_ = tanhf(acc[3][ms][v] + bo);
            const size_t idx = (size_t)row * UNITS + col;
            const float cn = f_ * c_state[idx] + i_ * g_;
            const float hn = o_ * tanhf(cn);
            c_state[idx] = cn;
            hnew[idx]    = (__bf16)hn;
            if (y_out) {
                const int hsp = col >> 9;      // /512
                const int d   = col & 511;
                y_out[(((size_t)row * 4 + hsp) * 4 + w_idx) * SIZE + d] = hn;
            }
        }
    }
}

// ===========================================================================
// Instance-norm(H,W) + LeakyReLU + transpose into LSTM step layout.
// H: [B,4,4,512] f32 (row = (b*4+h)*4+w) -> steps bf16 [B, t=W, h*512+d]
// ===========================================================================
__global__ __launch_bounds__(256) void in_lrelu_steps_kernel(
    const float* __restrict__ H, const float* __restrict__ gamma,
    const float* __restrict__ beta, __bf16* __restrict__ steps)
{
    const int t = blockIdx.x * blockDim.x + threadIdx.x;  // 0..131071
    const int b = t >> 9;
    const int d = t & 511;

    float v[16];
    float s = 0.f;
#pragma unroll
    for (int i = 0; i < 16; ++i) {
        v[i] = H[((size_t)b * 16 + i) * SIZE + d];
        s += v[i];
    }
    const float mu = s * (1.f / 16.f);
    float var = 0.f;
#pragma unroll
    for (int i = 0; i < 16; ++i) { float dv = v[i] - mu; var += dv * dv; }
    var *= (1.f / 16.f);
    const float inv = rsqrtf(var + EPS_IN);
    const float ga = gamma[d], be = beta[d];
#pragma unroll
    for (int i = 0; i < 16; ++i) {
        const int hsp = i >> 2, w = i & 3;
        float xn = (v[i] - mu) * inv * ga + be;
        xn = (xn >= 0.f) ? xn : LRELU_A * xn;
        steps[((size_t)b * 4 + w) * UNITS + hsp * SIZE + d] = (__bf16)xn;
    }
}

// ===========================================================================
// Final instance-norm + LeakyReLU on y2 [B,4,4,1024] f32 -> d_out f32.
// ===========================================================================
__global__ __launch_bounds__(256) void in_lrelu_out_kernel(
    const float* __restrict__ Y, const float* __restrict__ gamma,
    const float* __restrict__ beta, float* __restrict__ out)
{
    const int t = blockIdx.x * blockDim.x + threadIdx.x;  // 0..262143
    const int b = t >> 10;
    const int ch = t & 1023;

    float v[16];
    float s = 0.f;
#pragma unroll
    for (int i = 0; i < 16; ++i) {
        v[i] = Y[((size_t)b * 16 + i) * CCH + ch];
        s += v[i];
    }
    const float mu = s * (1.f / 16.f);
    float var = 0.f;
#pragma unroll
    for (int i = 0; i < 16; ++i) { float dv = v[i] - mu; var += dv * dv; }
    var *= (1.f / 16.f);
    const float inv = rsqrtf(var + EPS_IN);
    const float ga = gamma[ch], be = beta[ch];
#pragma unroll
    for (int i = 0; i < 16; ++i) {
        float xn = (v[i] - mu) * inv * ga + be;
        xn = (xn >= 0.f) ? xn : LRELU_A * xn;
        out[((size_t)b * 16 + i) * CCH + ch] = xn;
    }
}

__global__ __launch_bounds__(256) void zero_kernel(uint32_t* __restrict__ p, size_t n) {
    const size_t i = (size_t)blockIdx.x * blockDim.x + threadIdx.x;
    if (i < n) p[i] = 0u;
}

// ===========================================================================
// Host-side orchestration
// ===========================================================================
extern "C" void kernel_launch(void* const* d_in, const int* in_sizes, int n_in,
                              void* d_out, int out_size, void* d_ws, size_t ws_size,
                              hipStream_t stream)
{
    (void)in_sizes; (void)n_in; (void)out_size;

    const float* x        = (const float*)d_in[0];   // [256,4,4,1024]
    const float* conv1_w  = (const float*)d_in[1];   // [1024,512]
    const float* in1_g    = (const float*)d_in[2];
    const float* in1_b    = (const float*)d_in[3];
    const float* enc_k    = (const float*)d_in[4];   // [2048,8192]
    const float* enc_r    = (const float*)d_in[5];
    const float* enc_b    = (const float*)d_in[6];
    const float* dec_k    = (const float*)d_in[7];
    const float* dec_r    = (const float*)d_in[8];
    const float* dec_b    = (const float*)d_in[9];
    const float* conv2_w  = (const float*)d_in[10];  // [512,1024]
    const float* in2_g    = (const float*)d_in[11];
    const float* in2_b    = (const float*)d_in[12];
    float* out = (float*)d_out;

    // ---- workspace layout ----
    char* base = (char*)d_ws;
    size_t off = 0;
    auto alloc = [&](size_t bytes) { char* p = base + off; off += (bytes + 255) & ~(size_t)255; return p; };

    float*  H      = (float*) alloc((size_t)4096 * 512 * 4);   // conv1 out
    __bf16* steps  = (__bf16*)alloc((size_t)256 * 4 * 2048 * 2);
    // state buffers (contiguous for one-shot zeroing)
    char*   state0 = base + off;
    float*  c1     = (float*) alloc((size_t)256 * 2048 * 4);
    float*  c2     = (float*) alloc((size_t)256 * 2048 * 4);
    __bf16* h1[2]; __bf16* h2[2];
    h1[0] = (__bf16*)alloc((size_t)256 * 2048 * 2);
    h1[1] = (__bf16*)alloc((size_t)256 * 2048 * 2);
    h2[0] = (__bf16*)alloc((size_t)256 * 2048 * 2);
    h2[1] = (__bf16*)alloc((size_t)256 * 2048 * 2);
    size_t state_bytes = (size_t)(base + off - state0);
    float*  y_f32  = (float*) alloc((size_t)4096 * 512 * 4);   // decoder out, conv2 A
    float*  y2     = (float*) alloc((size_t)4096 * 1024 * 4);  // conv2 out
    if (off > ws_size) return;  // workspace too small; avoid corruption

    // ---- conv1: [4096,1024] x [1024,512] ----
    gemm_f32_bf16wmma<<<dim3(4096 / GM_BM, 512 / GM_BN), 256, 0, stream>>>(
        x, conv1_w, H, 4096, 512, 1024);

    // ---- instance norm + lrelu + transpose -> steps ----
    in_lrelu_steps_kernel<<<(256 * 512) / 256, 256, 0, stream>>>(H, in1_g, in1_b, steps);

    // ---- zero LSTM state ----
    {
        size_t ndw = state_bytes / 4;
        zero_kernel<<<(unsigned)((ndw + 255) / 256), 256, 0, stream>>>((uint32_t*)state0, ndw);
    }

    int p1 = 0, p2 = 0;
    auto cell = [&](const __bf16* xt, int xs, int layer,
                    const float* Kw, const float* Rw, const float* bias,
                    float* y, int w) {
        __bf16 *hin, *hout; float* c;
        if (layer == 1) { hin = h1[p1]; hout = h1[p1 ^ 1]; c = c1; p1 ^= 1; }
        else            { hin = h2[p2]; hout = h2[p2 ^ 1]; c = c2; p2 ^= 1; }
        lstm_cell_kernel<<<dim3(256 / LC_BM, 2048 / LC_BN), 256, 0, stream>>>(
            xt, xs, hin, Kw, Rw, bias, c, hout, y, w);
    };

    // ---- encoder: 4 steps, 2 layers sharing enc weights ----
    for (int t = 0; t < 4; ++t) {
        cell(steps + (size_t)t * 2048, 4 * 2048, 1, enc_k, enc_r, enc_b, nullptr, 0);
        cell(h1[p1],                   2048,     2, enc_k, enc_r, enc_b, nullptr, 0);
    }
    // ---- decoder: feed back h2; scatter layer-2 outputs into y ----
    for (int s = 0; s < 4; ++s) {
        const __bf16* now = (s == 0) ? steps + (size_t)3 * 2048 : h2[p2];
        const int xs      = (s == 0) ? 4 * 2048 : 2048;
        cell(now,    xs,   1, dec_k, dec_r, dec_b, nullptr, 0);
        cell(h1[p1], 2048, 2, dec_k, dec_r, dec_b, y_f32, s);
    }

    // ---- conv2: [4096,512] x [512,1024] ----
    gemm_f32_bf16wmma<<<dim3(4096 / GM_BM, 1024 / GM_BN), 256, 0, stream>>>(
        y_f32, conv2_w, y2, 4096, 1024, 512);

    // ---- final instance norm + lrelu ----
    in_lrelu_out_kernel<<<(256 * 1024) / 256, 256, 0, stream>>>(y2, in2_g, in2_b, out);
}
